// MaskConv_87308095193368
// MI455X (gfx1250) — matrix operand order repaired and checked by reading, not compiled, not run on hardware
//
#include <hip/hip_runtime.h>
#include <hip/hip_bf16.h>

typedef __attribute__((ext_vector_type(16))) __bf16 v16bf;
typedef __attribute__((ext_vector_type(8)))  float  v8f;
typedef __bf16 v8bf __attribute__((ext_vector_type(8), aligned(2)));   // unaligned-capable

// ---- geometry ------------------------------------------------------------
// x  : [16, 1, 161, 1024] f32
// xp : [16, 202, 1040]    bf16 padded input   (conv1 halo + dummy row 41)
// h1p: [16, 32, 102, 528] bf16 padded hidden  (conv2 halo + dummy row 21 + kw' slack)
// out: [16, 32, 41, 512]  f32
//
// K reordered as (row, kw'), kw' padded 11->16, kh padded so rows per slice is even:
//   conv1: 42 rows (41 real + 1 zero)            -> K1R = 672   = 21 chunks
//   conv2: 32 ci x 22 rows (21 real + 1 zero)    -> K2R = 11264 = 32 x 11 chunks
#define K1R     672
#define NC1     21
#define K2R     11264
#define XP_W    1040
#define XP_IMG  (202 * 1040)          // 210080
#define HP_W    528
#define HP_PLANE (102 * 528)          // 53856
#define HP_IMG  (32 * HP_PLANE)       // 1723392

// workspace layout (bytes, 256B-aligned sections)
#define WS_H1P  0u
#define WS_XP   55148544u                       // 16*HP_IMG*2
#define WS_W1   (WS_XP + 6722560u)              // 16*XP_IMG*2
#define WS_W2   (WS_W1 + 43008u)                // 32*K1R*2
#define WS_BN   (WS_W2 + 720896u)               // 32*K2R*2 ; bn = 128 f32

__device__ __forceinline__ v16bf cat8(v8bf a, v8bf b)
{
    return __builtin_shufflevector(a, b, 0, 1, 2, 3, 4, 5, 6, 7,
                                         8, 9, 10, 11, 12, 13, 14, 15);
}
__device__ __forceinline__ v8f bfwmma(v16bf a, v16bf b, v8f c)
{
    return __builtin_amdgcn_wmma_f32_16x16x32_bf16(false, a, false, b,
                                                   (short)0, c, false, false);
}
__device__ __forceinline__ float htanh(float v) { return fminf(fmaxf(v, 0.0f), 20.0f); }

// ---- prep: fold BN/bias, build reordered bf16 weights --------------------
__global__ void prep_kernel(const float* __restrict__ w1, const float* __restrict__ w2,
                            const float* __restrict__ b1, const float* __restrict__ g1,
                            const float* __restrict__ be1, const float* __restrict__ m1,
                            const float* __restrict__ v1,
                            const float* __restrict__ b2, const float* __restrict__ g2,
                            const float* __restrict__ be2, const float* __restrict__ m2,
                            const float* __restrict__ v2,
                            __bf16* __restrict__ w1r, __bf16* __restrict__ w2r,
                            float* __restrict__ bn)
{
    const int i = blockIdx.x * blockDim.x + threadIdx.x;
    if (i < 32) {
        float s1 = g1[i] * rsqrtf(v1[i] + 1e-5f);
        bn[i]      = s1;
        bn[32 + i] = (b1[i] - m1[i]) * s1 + be1[i];
        float s2 = g2[i] * rsqrtf(v2[i] + 1e-5f);
        bn[64 + i] = s2;
        bn[96 + i] = (b2[i] - m2[i]) * s2 + be2[i];
    }
    if (i < 32 * K1R) {      // [c][kh'(42)][kw'(16)] ; real: kh<41 && kw<11
        int c  = i / K1R, k = i - c * K1R;
        int kh = k >> 4, kw = k & 15;
        float v = (kh < 41 && kw < 11) ? w1[c * 451 + kh * 11 + kw] : 0.0f;
        w1r[i] = (__bf16)v;
    }
    if (i < 32 * K2R) {      // [c][ci(32)][kh'(22)][kw'(16)] ; real: kh<21 && kw<11
        int c   = i / K2R, k = i - c * K2R;
        int row = k >> 4, kw = k & 15;
        int ci  = row / 22, kh = row - ci * 22;
        float v = (kh < 21 && kw < 11) ? w2[(c * 32 + ci) * 231 + kh * 11 + kw] : 0.0f;
        w2r[i] = (__bf16)v;
    }
}

// ---- zero the padded hidden buffer ---------------------------------------
__global__ void zero_h1p_kernel(uint4* __restrict__ p, int n16)
{
    const int i = blockIdx.x * blockDim.x + threadIdx.x;
    if (i < n16) p[i] = make_uint4(0u, 0u, 0u, 0u);
}

// ---- pad + convert input to bf16 -----------------------------------------
__global__ void pad_x_kernel(const float* __restrict__ x, __bf16* __restrict__ xp)
{
    const int i = blockIdx.x * blockDim.x + threadIdx.x;
    if (i >= 16 * XP_IMG) return;
    const int n  = i / XP_IMG;
    const int r  = i - n * XP_IMG;
    const int rp = r / XP_W;
    const int cp = r - rp * XP_W;
    const int h  = rp - 20;
    const int w  = cp - 5;
    float v = 0.0f;
    if ((unsigned)h < 161u && (unsigned)w < 1024u) v = x[(n * 161 + h) * 1024 + w];
    xp[i] = (__bf16)v;
}

// ---- conv1 + BN + hardtanh + mask : 32x32 tile / wave, all-immediate -----
__global__ __launch_bounds__(256)
void conv1_wmma(const __bf16* __restrict__ xp, const int* __restrict__ lens,
                const __bf16* __restrict__ wb, const float* __restrict__ bn,
                __bf16* __restrict__ h1p)
{
    const int lane   = threadIdx.x & 31;
    const int wave   = threadIdx.x >> 5;
    const int col    = lane & 15;
    const int half   = lane >> 4;
    const int m_base = blockIdx.x * 256 + wave * 32;

    const int n   = m_base / (81 * 512);
    const int rm  = m_base - n * (81 * 512);
    const int ho  = rm >> 9;
    const int wob = rm & 511;

    const __bf16* paLo = xp + n * XP_IMG + (2 * ho) * XP_W + 2 * (wob + col) + 8 * half;
    const __bf16* paHi = paLo + 32;                 // +16 positions, stride 2
    const __bf16* pb0  = wb + col * K1R + half * 8;
    const __bf16* pb1  = pb0 + 16 * K1R;

    v8f a00 = {}, a01 = {}, a10 = {}, a11 = {};
    #pragma unroll
    for (int cc = 0; cc < NC1; ++cc) {              // all offsets are immediates
        const int ra = (2 * cc) * XP_W;
        const int rb = ra + XP_W;
        v16bf aLo = cat8(*(const v8bf*)(paLo + ra), *(const v8bf*)(paLo + rb));
        v16bf aHi = cat8(*(const v8bf*)(paHi + ra), *(const v8bf*)(paHi + rb));
        v16bf b0  = cat8(*(const v8bf*)(pb0 + cc * 32), *(const v8bf*)(pb0 + cc * 32 + 16));
        v16bf b1  = cat8(*(const v8bf*)(pb1 + cc * 32), *(const v8bf*)(pb1 + cc * 32 + 16));
        a00 = bfwmma(aLo, b0, a00);  a01 = bfwmma(aLo, b1, a01);
        a10 = bfwmma(aHi, b0, a10);  a11 = bfwmma(aHi, b1, a11);
    }

    const int   len_n = lens[n];
    const float s0 = bn[col],      t0 = bn[32 + col];
    const float s1 = bn[col + 16], t1 = bn[32 + col + 16];
    __bf16* o0 = h1p + (long)n * HP_IMG + (col)      * HP_PLANE + (ho + 10) * HP_W + 5;
    __bf16* o1 = h1p + (long)n * HP_IMG + (col + 16) * HP_PLANE + (ho + 10) * HP_W + 5;
    #pragma unroll
    for (int r = 0; r < 8; ++r) {
        const int  wA = wob + 8 * half + r;
        const int  wB = wA + 16;
        const bool kA = wA < len_n, kB = wB < len_n;
        o0[wA] = (__bf16)(kA ? htanh(a00[r] * s0 + t0) : 0.0f);
        o1[wA] = (__bf16)(kA ? htanh(a01[r] * s1 + t1) : 0.0f);
        o0[wB] = (__bf16)(kB ? htanh(a10[r] * s0 + t0) : 0.0f);
        o1[wB] = (__bf16)(kB ? htanh(a11[r] * s1 + t1) : 0.0f);
    }
}

// ---- conv2 + BN + hardtanh + mask : 32x32 tile / wave --------------------
__global__ __launch_bounds__(256)
void conv2_wmma(const __bf16* __restrict__ h1p, const int* __restrict__ lens,
                const __bf16* __restrict__ wb, const float* __restrict__ bn,
                float* __restrict__ out)
{
    const int lane   = threadIdx.x & 31;
    const int wave   = threadIdx.x >> 5;
    const int col    = lane & 15;
    const int half   = lane >> 4;
    const int m_base = blockIdx.x * 256 + wave * 32;

    const int n   = m_base / (41 * 512);
    const int rm  = m_base - n * (41 * 512);
    const int ho  = rm >> 9;
    const int wob = rm & 511;

    const __bf16* paLo = h1p + (long)n * HP_IMG + (2 * ho) * HP_W + (wob + col) + 8 * half;
    const __bf16* paHi = paLo + 16;                 // +16 positions, stride 1
    const __bf16* pb0  = wb + col * K2R + half * 8;
    const __bf16* pb1  = pb0 + 16 * K2R;

    v8f a00 = {}, a01 = {}, a10 = {}, a11 = {};
    for (int ci = 0; ci < 32; ++ci) {
        #pragma unroll
        for (int cc = 0; cc < 11; ++cc) {           // all offsets are immediates
            const int ra = (2 * cc) * HP_W;
            const int rb = ra + HP_W;
            v16bf aLo = cat8(*(const v8bf*)(paLo + ra), *(const v8bf*)(paLo + rb));
            v16bf aHi = cat8(*(const v8bf*)(paHi + ra), *(const v8bf*)(paHi + rb));
            v16bf b0  = cat8(*(const v8bf*)(pb0 + cc * 32), *(const v8bf*)(pb0 + cc * 32 + 16));
            v16bf b1  = cat8(*(const v8bf*)(pb1 + cc * 32), *(const v8bf*)(pb1 + cc * 32 + 16));
            a00 = bfwmma(aLo, b0, a00);  a01 = bfwmma(aLo, b1, a01);
            a10 = bfwmma(aHi, b0, a10);  a11 = bfwmma(aHi, b1, a11);
        }
        paLo += HP_PLANE; paHi += HP_PLANE;         // next input channel slice
        pb0  += 352;      pb1  += 352;              // 11 chunks * 32
    }

    const int   len_n = lens[n];
    const float s0 = bn[col],      t0 = bn[32 + col];
    const float s1 = bn[col + 16], t1 = bn[32 + col + 16];
    float* o0 = out + ((n * 32 + col)      * 41 + ho) * 512;
    float* o1 = out + ((n * 32 + col + 16) * 41 + ho) * 512;
    #pragma unroll
    for (int r = 0; r < 8; ++r) {
        const int  wA = wob + 8 * half + r;
        const int  wB = wA + 16;
        const bool kA = wA < len_n, kB = wB < len_n;
        o0[wA] = kA ? htanh(a00[r] * s0 + t0) : 0.0f;
        o1[wA] = kA ? htanh(a01[r] * s1 + t1) : 0.0f;
        o0[wB] = kB ? htanh(a10[r] * s0 + t0) : 0.0f;
        o1[wB] = kB ? htanh(a11[r] * s1 + t1) : 0.0f;
    }
}

// ---- host launch ---------------------------------------------------------
extern "C" void kernel_launch(void* const* d_in, const int* in_sizes, int n_in,
                              void* d_out, int out_size, void* d_ws, size_t ws_size,
                              hipStream_t stream)
{
    (void)in_sizes; (void)n_in; (void)out_size; (void)ws_size;

    const float* x    = (const float*)d_in[0];
    const int*   lens = (const int*)  d_in[1];
    const float* w1   = (const float*)d_in[2];
    const float* b1   = (const float*)d_in[3];
    const float* g1   = (const float*)d_in[4];
    const float* be1  = (const float*)d_in[5];
    const float* m1   = (const float*)d_in[6];
    const float* v1   = (const float*)d_in[7];
    const float* w2   = (const float*)d_in[8];
    const float* b2   = (const float*)d_in[9];
    const float* g2   = (const float*)d_in[10];
    const float* be2  = (const float*)d_in[11];
    const float* m2   = (const float*)d_in[12];
    const float* v2   = (const float*)d_in[13];

    char*   ws  = (char*)d_ws;
    __bf16* h1p = (__bf16*)(ws + WS_H1P);
    __bf16* xp  = (__bf16*)(ws + WS_XP);
    __bf16* w1r = (__bf16*)(ws + WS_W1);
    __bf16* w2r = (__bf16*)(ws + WS_W2);
    float*  bn  = (float*) (ws + WS_BN);

    prep_kernel<<<(32 * K2R + 255) / 256, 256, 0, stream>>>(
        w1, w2, b1, g1, be1, m1, v1, b2, g2, be2, m2, v2, w1r, w2r, bn);

    const int n16 = (int)(55148544u / 16u);
    zero_h1p_kernel<<<(n16 + 255) / 256, 256, 0, stream>>>((uint4*)h1p, n16);

    pad_x_kernel<<<(16 * XP_IMG + 255) / 256, 256, 0, stream>>>(x, xp);

    // conv1: M = 16*81*512 = 663552 positions, 256 per block (8 waves x 32)
    conv1_wmma<<<663552 / 256, 256, 0, stream>>>(xp, lens, w1r, bn, h1p);

    // conv2: M = 16*41*512 = 335872 positions
    conv2_wmma<<<335872 / 256, 256, 0, stream>>>(h1p, lens, w2r, bn + 64,
                                                 (float*)d_out);
}
